// ContextTransformer_24266565222640
// MI455X (gfx1250) — matrix operand order repaired and checked by reading, not compiled, not run on hardware
//
#include <hip/hip_runtime.h>

// ---------------------------------------------------------------------------
// CDNA5 (gfx1250) ContextTransformer forward.
// All dense GEMMs run through v_wmma_f32_16x16x32_bf16 (wave32 WMMA) with
// double-buffered LDS weight tiles (ds_store_b128/ds_load_b128, one barrier
// per K-step) and register double-buffering so global latency overlaps the
// WMMA pipe. Weight-bandwidth-bound workload: weights stream exactly once.
// ---------------------------------------------------------------------------

typedef __bf16 bf16;
typedef __attribute__((ext_vector_type(16))) __bf16 v16bf;
typedef __attribute__((ext_vector_type(8)))  __bf16 v8bf;
typedef __attribute__((ext_vector_type(8)))  float  v8f;

constexpr int kB = 32, kT = 192, kM = 256, kP = 16;
constexpr int kD = 1024, kHE = 1024, kHD = 1024, kHG = 512;
constexpr int kNH = 16, kDH = 64, kDFF = 4096, kNL = 8;
constexpr int kCTX = 10, kE = 8;
constexpr int kBT = kB * kT;        // 6144 tokens
constexpr int kREL = 2 * kT - 1;    // 383 rel-pos rows

__device__ __forceinline__ unsigned short f32_to_bf16(float f) {
  unsigned u = __builtin_bit_cast(unsigned, f);
  unsigned r = 0x7FFFu + ((u >> 16) & 1u);   // round-to-nearest-even
  return (unsigned short)((u + r) >> 16);
}
__device__ __forceinline__ float preluf(float v) { return v >= 0.f ? v : 0.25f * v; }

// ------------------------- conversion kernels ------------------------------

// fp32 [rows,K] -> bf16 [rows,Kpad] (zero pad tail so WMMA K-loop is exact)
__global__ __launch_bounds__(256)
void cvt_pad_rows(const float* __restrict__ src, unsigned short* __restrict__ dst,
                  int rows, int K, int Kpad) {
  int i = blockIdx.x * 256 + threadIdx.x;
  int r = i / Kpad, c = i - r * Kpad;
  if (r >= rows) return;
  float v = (c < K) ? src[(size_t)r * K + c] : 0.f;
  dst[i] = f32_to_bf16(v);
}

// fp32 weight [K,N] -> bf16 transposed [N,Kpad] (so B fragments load contiguously)
__global__ __launch_bounds__(256)
void cvt_w_transpose(const float* __restrict__ w, unsigned short* __restrict__ dst,
                     int K, int N, int Kpad) {
  int i = blockIdx.x * 256 + threadIdx.x;
  int n = i / Kpad, c = i - n * Kpad;
  if (n >= N) return;
  float v = (c < K) ? w[(size_t)c * N + n] : 0.f;
  dst[i] = f32_to_bf16(v);
}

// Build masked encoder input [BT, 273] directly as padded bf16 [BT, 288]
__global__ __launch_bounds__(256)
void build_x0_bf16(const float* __restrict__ motion, const float* __restrict__ phase,
                   unsigned short* __restrict__ dst) {
  int i = blockIdx.x * 256 + threadIdx.x;
  const int Kpad = 288;
  int r = i / Kpad, c = i - r * Kpad;
  if (r >= kBT) return;
  int t = r % kT;
  float dm = (t < kCTX || t == kT - 1) ? 1.f : 0.f;
  float v = 0.f;
  if (c < kM)            v = motion[(size_t)r * kM + c] * dm;
  else if (c == kM)      v = dm;
  else if (c < kM + 1 + kP) v = phase[(size_t)r * kP + (c - kM - 1)] * dm;
  dst[i] = f32_to_bf16(v);
}

// ------------------------------ WMMA GEMM ----------------------------------
// C[M,N] = epi( sum_k A[m,k]*Bt[n,k] )  with A,Bt bf16 row-major [*,Kpad].
// Block = 256 threads = 8 waves; wave w -> rows [blockX*256 + 32w, +32)
// (two A fragments), cols [blockY*64, +64) as 4 subtiles -> 8 WMMAs/K-step.
// B tile (64x32) staged through double-buffered LDS: store next stage while
// computing current, ONE barrier per K-step. A fragments + B stage chunk are
// register-double-buffered so the global s_wait lands after the WMMAs.
// act: 0=none 1=PReLU(0.25) 2=ReLU.  rowscale (stride rstride) = MoE gate.
// accum: C += val (residual / expert accumulation) else C = val.
union Frag { v16bf v; v8bf h[2]; };

constexpr int kLDSROW = 40;   // bf16 per LDS row (80B, padded from 64B):
                              // row*20 words, de-staggered bank quads

__global__ __launch_bounds__(256)
void gemm_wmma_bf16(const bf16* __restrict__ A, const bf16* __restrict__ Bt,
                    const float* __restrict__ bias, const float* __restrict__ rowscale,
                    int rstride, float* __restrict__ C,
                    int M, int N, int Kpad, int act, int accum) {
  __shared__ bf16 sB[2][64 * kLDSROW];       // 2 x 5KB weight staging tiles
  const int tid = threadIdx.x;
  const int wave = tid >> 5, lane = tid & 31;
  const int m0 = blockIdx.x * 256 + wave * 32;
  const int n0 = blockIdx.y * 64;
  const int half = lane >> 4, lrow = lane & 15;

  // ---- B staging addressing: thread t fetches 16B of tile row (t>>2), seg (t&3)
  const int srow = tid >> 2, sseg = tid & 3;
  int gn = n0 + srow; if (gn >= N) gn = N - 1;
  const bf16* gB  = Bt + (size_t)gn * Kpad + sseg * 8;
  const int  sOff = srow * kLDSROW + sseg * 8;

  // ---- A fragment addressing (ISA 16-bit A 16x32): lanes 0-15 hold K 0-7 &
  // 16-23, lanes 16-31 hold K 8-15 & 24-31 of row M = lane&15. Rows clamped
  // for out-of-range waves (they skip the epilogue store).
  int ar0 = m0 + lrow;      if (ar0 >= M) ar0 = M - 1;
  int ar1 = m0 + 16 + lrow; if (ar1 >= M) ar1 = M - 1;
  const bf16* aptr0 = A + (size_t)ar0 * Kpad + half * 8;
  const bf16* aptr1 = A + (size_t)ar1 * Kpad + half * 8;

  // ---- B fragment LDS read base: subtile j, column lane&15, K (l>>4)*8 & +16
  const int ldOff = lrow * kLDSROW + half * 8;

  const v8f zero = {0.f, 0.f, 0.f, 0.f, 0.f, 0.f, 0.f, 0.f};
  v8f acc[2][4] = {{zero, zero, zero, zero}, {zero, zero, zero, zero}};

  // ---- prologue: stage k=0
  v8bf stg = *(const v8bf*)(gB);
  Frag a0, a1;
  a0.h[0] = *(const v8bf*)(aptr0);
  a0.h[1] = *(const v8bf*)(aptr0 + 16);
  a1.h[0] = *(const v8bf*)(aptr1);
  a1.h[1] = *(const v8bf*)(aptr1 + 16);
  *(v8bf*)(&sB[0][0] + sOff) = stg;
  __syncthreads();

  const int nk = Kpad >> 5;
  int s = 0;
  for (int kt = 0; kt < nk; ++kt) {
    const int kn = (kt + 1) << 5;
    const bool more = (kt + 1 < nk);
    Frag a0n, a1n; v8bf stgN;
    if (more) {                               // issue next-tile globals early
      __builtin_prefetch(gB + kn + 128, 0, 3);
      __builtin_prefetch(aptr0 + kn + 128, 0, 3);
      stgN      = *(const v8bf*)(gB + kn);
      a0n.h[0] = *(const v8bf*)(aptr0 + kn);
      a0n.h[1] = *(const v8bf*)(aptr0 + kn + 16);
      a1n.h[0] = *(const v8bf*)(aptr1 + kn);
      a1n.h[1] = *(const v8bf*)(aptr1 + kn + 16);
    }
    // B fragments from LDS (ISA 16-bit B 32x16: same K order as A fragments)
    const bf16* base = &sB[s][0] + ldOff;
#pragma unroll
    for (int j = 0; j < 4; ++j) {
      Frag fb;
      fb.h[0] = *(const v8bf*)(base + j * 16 * kLDSROW);
      fb.h[1] = *(const v8bf*)(base + j * 16 * kLDSROW + 16);
      acc[0][j] = __builtin_amdgcn_wmma_f32_16x16x32_bf16(
          false, a0.v, false, fb.v, (short)0, acc[0][j], false, false);
      acc[1][j] = __builtin_amdgcn_wmma_f32_16x16x32_bf16(
          false, a1.v, false, fb.v, (short)0, acc[1][j], false, false);
    }
    // write the *other* stage while this one is being read; reads of stage
    // s^1 from iteration kt-1 were fenced by that iteration's barrier.
    if (more) *(v8bf*)(&sB[s ^ 1][0] + sOff) = stgN;
    __syncthreads();                          // single barrier per K-step
    if (more) { a0 = a0n; a1 = a1n; }
    s ^= 1;
  }

  // C layout: lane l -> column n0 + (l&15); VGPR r -> row m0 + r + 8*(l>>4)
#pragma unroll
  for (int i = 0; i < 2; ++i) {
#pragma unroll
    for (int j = 0; j < 4; ++j) {
      int n = n0 + j * 16 + lrow;
      if (n >= N) continue;
      float bb = bias ? bias[n] : 0.f;
#pragma unroll
      for (int r = 0; r < 8; ++r) {
        int m = m0 + i * 16 + half * 8 + r;
        if (m >= M) continue;
        float val = acc[i][j][r] + bb;
        if (act == 1)      val = preluf(val);
        else if (act == 2) val = fmaxf(val, 0.f);
        if (rowscale) val *= rowscale[(size_t)m * rstride];
        float* cp = C + (size_t)m * N + n;
        if (accum) *cp += val; else *cp = val;
      }
    }
  }
}

// ------------------------------ LayerNorm ----------------------------------
__global__ __launch_bounds__(256)
void layernorm_k(const float* __restrict__ x, const float* __restrict__ g,
                 const float* __restrict__ b, float* __restrict__ out, int D) {
  __shared__ float red[256];
  const int row = blockIdx.x, tid = threadIdx.x;
  const float* xr = x + (size_t)row * D;
  float s = 0.f;
  for (int i = tid; i < D; i += 256) s += xr[i];
  red[tid] = s; __syncthreads();
  for (int k = 128; k; k >>= 1) { if (tid < k) red[tid] += red[tid + k]; __syncthreads(); }
  const float mean = red[0] / D;
  __syncthreads();
  float v = 0.f;
  for (int i = tid; i < D; i += 256) { float d = xr[i] - mean; v += d * d; }
  red[tid] = v; __syncthreads();
  for (int k = 128; k; k >>= 1) { if (tid < k) red[tid] += red[tid + k]; __syncthreads(); }
  const float rstd = rsqrtf(red[0] / D + 1e-5f);
  __syncthreads();
  float* orow = out + (size_t)row * D;
  for (int i = tid; i < D; i += 256) orow[i] = (xr[i] - mean) * rstd * g[i] + b[i];
}

// ------------------------------ Attention ----------------------------------
// grid: (B*NH, T/8); one wave per query row. wave32 softmax via __shfl_xor.
__global__ __launch_bounds__(256)
void attention_k(const float* __restrict__ q, const float* __restrict__ ke,
                 const float* __restrict__ ve, const float* __restrict__ rel,
                 float* __restrict__ o) {
  __shared__ float probs[8][kT];
  const int wave = threadIdx.x >> 5, lane = threadIdx.x & 31;
  const int bh = blockIdx.x;
  const int b = bh / kNH, hd = bh - b * kNH;
  const int qi = blockIdx.y * 8 + wave;
  const size_t stride = (size_t)kNH * kDH;
  const float* qrow = q + ((size_t)(b * kT + qi)) * stride + hd * kDH;
  const float scale = 0.125f;                  // 1/sqrt(DH)

  float s[6], mx = -3.0e38f;
#pragma unroll
  for (int c = 0; c < 6; ++c) {
    const int j = c * 32 + lane;
    const float* krow = ke + ((size_t)(b * kT + j)) * stride + hd * kDH;
    const float* rrow = rel + (size_t)(j - qi + kT - 1) * kDH;
    float acc = 0.f;
    for (int d = 0; d < kDH; ++d) acc += qrow[d] * (krow[d] + rrow[d]);
    float sv = acc * scale;
    if (j >= kCTX && j != kT - 1) sv = -1e9f;  // attend only to keyframes
    s[c] = sv; mx = fmaxf(mx, sv);
  }
  for (int off = 16; off; off >>= 1) mx = fmaxf(mx, __shfl_xor(mx, off, 32));
  float sum = 0.f;
#pragma unroll
  for (int c = 0; c < 6; ++c) { s[c] = __expf(s[c] - mx); sum += s[c]; }
  for (int off = 16; off; off >>= 1) sum += __shfl_xor(sum, off, 32);
  const float inv = 1.f / sum;
#pragma unroll
  for (int c = 0; c < 6; ++c) probs[wave][c * 32 + lane] = s[c] * inv;
  __syncthreads();

  float o0 = 0.f, o1 = 0.f;
  const float* vbase = ve + ((size_t)(b * kT)) * stride + hd * kDH;
  for (int j = 0; j < kT; ++j) {
    const float p = probs[wave][j];
    const float* vrow = vbase + (size_t)j * stride;
    o0 += p * vrow[lane];
    o1 += p * vrow[lane + 32];
  }
  float* orow = o + ((size_t)(b * kT + qi)) * stride + hd * kDH;
  orow[lane] = o0; orow[lane + 32] = o1;
}

// ---------------------------- small kernels --------------------------------
__global__ __launch_bounds__(256)
void kf_hidden_k(const float* __restrict__ W1, const float* __restrict__ b1,
                 float* __restrict__ out) {
  int i = blockIdx.x * 256 + threadIdx.x;
  if (i >= kT * kD) return;
  int t = i / kD, d = i - t * kD;
  float ctx = (float)(t - (kCTX - 1));
  float tgt = (float)(kT - 1 - t);
  out[i] = preluf(ctx * W1[d] + tgt * W1[kD + d] + b1[d]);
}

__global__ __launch_bounds__(256)
void add_bcast_k(float* __restrict__ x, const float* __restrict__ kfe) {
  int i = blockIdx.x * 256 + threadIdx.x;
  if (i >= kBT * kD) return;
  int row = i / kD, col = i - row * kD;
  x[i] += kfe[(size_t)(row % kT) * kD + col];
}

__global__ __launch_bounds__(256)
void relh_k(const float* __restrict__ W1, const float* __restrict__ b1,
            float* __restrict__ out) {
  int i = blockIdx.x * 256 + threadIdx.x;
  if (i >= kREL * kDH) return;
  int r = i >> 6, d = i & 63;
  float rp = (float)(r - (kT - 1));
  out[i] = preluf(rp * W1[d] + b1[d]);
}

__global__ __launch_bounds__(256)
void rel2_k(const float* __restrict__ hsrc, const float* __restrict__ W2,
            const float* __restrict__ b2, float* __restrict__ out) {
  int i = blockIdx.x * 256 + threadIdx.x;
  if (i >= kREL * kDH) return;
  int r = i >> 6, dd = i & 63;
  float acc = b2[dd];
  for (int d = 0; d < kDH; ++d) acc += hsrc[r * kDH + d] * W2[d * kDH + dd];
  out[i] = acc;
}

__global__ __launch_bounds__(256)
void softmax_rows_k(const float* __restrict__ logit, float* __restrict__ gate,
                    int rows) {
  int row = blockIdx.x * 256 + threadIdx.x;
  if (row >= rows) return;
  const float* in = logit + (size_t)row * kE;
  float mx = in[0];
  for (int e = 1; e < kE; ++e) mx = fmaxf(mx, in[e]);
  float sum = 0.f, ex[kE];
  for (int e = 0; e < kE; ++e) { ex[e] = __expf(in[e] - mx); sum += ex[e]; }
  float inv = 1.f / sum;
  for (int e = 0; e < kE; ++e) gate[(size_t)row * kE + e] = ex[e] * inv;
}

__global__ __launch_bounds__(256)
void prelu_k(float* __restrict__ p, int n) {
  int i = blockIdx.x * 256 + threadIdx.x;
  if (i < n) p[i] = preluf(p[i]);
}

// ------------------------------- driver ------------------------------------
extern "C" void kernel_launch(void* const* d_in, const int* in_sizes, int n_in,
                              void* d_out, int out_size, void* d_ws, size_t ws_size,
                              hipStream_t stream) {
  (void)in_sizes; (void)n_in; (void)out_size;
  const float* motion   = (const float*)d_in[0];
  const float* phase    = (const float*)d_in[1];
  const float* enc_W1   = (const float*)d_in[2];
  const float* enc_b1   = (const float*)d_in[3];
  const float* enc_W2   = (const float*)d_in[4];
  const float* enc_b2   = (const float*)d_in[5];
  const float* kf_W1    = (const float*)d_in[6];
  const float* kf_b1    = (const float*)d_in[7];
  const float* kf_W2    = (const float*)d_in[8];
  const float* kf_b2    = (const float*)d_in[9];
  const float* rp_W1    = (const float*)d_in[10];
  const float* rp_b1    = (const float*)d_in[11];
  const float* rp_W2    = (const float*)d_in[12];
  const float* rp_b2    = (const float*)d_in[13];
  const float* att_Wq   = (const float*)d_in[14];
  const float* att_bq   = (const float*)d_in[15];
  const float* att_Wk   = (const float*)d_in[16];
  const float* att_bk   = (const float*)d_in[17];
  const float* att_Wv   = (const float*)d_in[18];
  const float* att_bv   = (const float*)d_in[19];
  const float* att_Wo   = (const float*)d_in[20];
  const float* att_bo   = (const float*)d_in[21];
  const float* att_ln_g = (const float*)d_in[22];
  const float* att_ln_b = (const float*)d_in[23];
  const float* pff_W1   = (const float*)d_in[24];
  const float* pff_b1   = (const float*)d_in[25];
  const float* pff_W2   = (const float*)d_in[26];
  const float* pff_b2   = (const float*)d_in[27];
  const float* pff_ln_g = (const float*)d_in[28];
  const float* pff_ln_b = (const float*)d_in[29];
  const float* fin_ln_g = (const float*)d_in[30];
  const float* fin_ln_b = (const float*)d_in[31];
  const float* pd_W1    = (const float*)d_in[32];
  const float* pd_b1    = (const float*)d_in[33];
  const float* pd_W2    = (const float*)d_in[34];
  const float* pd_b2    = (const float*)d_in[35];
  const float* g_W1     = (const float*)d_in[36];
  const float* g_b1     = (const float*)d_in[37];
  const float* g_W2     = (const float*)d_in[38];
  const float* g_b2     = (const float*)d_in[39];
  const float* g_W3     = (const float*)d_in[40];
  const float* g_b3     = (const float*)d_in[41];
  const float* md_W0    = (const float*)d_in[42];
  const float* md_b0    = (const float*)d_in[43];
  const float* md_W1    = (const float*)d_in[44];
  const float* md_b1    = (const float*)d_in[45];
  const float* md_W2    = (const float*)d_in[46];
  const float* md_b2    = (const float*)d_in[47];

  // ---- workspace carve-out ----
  char* ws = (char*)d_ws;
  size_t off = 0;
  auto alloc = [&](size_t bytes) -> void* {
    off = (off + 255) & ~(size_t)255;
    void* p = ws + off; off += bytes; return p;
  };
  float* x    = (float*)alloc((size_t)kBT * kD * 4);
  float* h    = (float*)alloc((size_t)kBT * kD * 4);     // also gg1/gg2
  float* big  = (float*)alloc((size_t)kBT * kDFF * 4);   // q,k,v,o | ff | MoE y2
  float* y    = (float*)alloc((size_t)kBT * kD * 4);     // pd hidden | MoE y
  unsigned short* Abf = (unsigned short*)alloc((size_t)kBT * kDFF * 2);
  unsigned short* Btb = (unsigned short*)alloc((size_t)kDFF * kD * 2);
  float* kfh  = (float*)alloc((size_t)kT * kD * 4);
  float* kfe  = (float*)alloc((size_t)kT * kD * 4);
  float* relh = (float*)alloc((size_t)kREL * kDH * 4);
  float* relE = (float*)alloc((size_t)kREL * kDH * 4);
  float* glog = (float*)alloc((size_t)kBT * kE * 4);
  float* gate = (float*)alloc((size_t)kBT * kE * 4);
  if (off > ws_size) return;

  float* qb = big;
  float* kb = big + (size_t)kBT * kD;
  float* vb = big + (size_t)2 * kBT * kD;
  float* ob = big + (size_t)3 * kBT * kD;
  float* gg1 = h;
  float* gg2 = h + (size_t)kBT * kHG;
  float* yout      = (float*)d_out;                       // [BT, 256]
  float* phase_out = (float*)d_out + (size_t)kBT * kM;    // [BT, 16]

  auto cvtR = [&](const float* s, int rows, int K, int Kpad) {
    int n = rows * Kpad;
    cvt_pad_rows<<<(n + 255) / 256, 256, 0, stream>>>(s, Abf, rows, K, Kpad);
  };
  auto cvtW = [&](const float* w, int K, int N, int Kpad) {
    int n = N * Kpad;
    cvt_w_transpose<<<(n + 255) / 256, 256, 0, stream>>>(w, Btb, K, N, Kpad);
  };
  auto gemm = [&](const float* bias, const float* rowscale, int rstride, float* C,
                  int M, int N, int Kpad, int act, int accum) {
    dim3 g((M + 255) / 256, (N + 63) / 64);
    gemm_wmma_bf16<<<g, 256, 0, stream>>>((const bf16*)Abf, (const bf16*)Btb,
                                          bias, rowscale, rstride, C,
                                          M, N, Kpad, act, accum);
  };

  // ---- encoder ----
  { int n = kBT * 288;
    build_x0_bf16<<<(n + 255) / 256, 256, 0, stream>>>(motion, phase, Abf); }
  cvtW(enc_W1, kM + 1 + kP, kHE, 288);
  gemm(enc_b1, nullptr, 0, h, kBT, kHE, 288, 1, 0);
  cvtR(h, kBT, kHE, kHE);
  cvtW(enc_W2, kHE, kD, kHE);
  gemm(enc_b2, nullptr, 0, x, kBT, kD, kHE, 1, 0);

  // ---- keyframe position embedding ----
  { int n = kT * kD;
    kf_hidden_k<<<(n + 255) / 256, 256, 0, stream>>>(kf_W1, kf_b1, kfh); }
  cvtR(kfh, kT, kD, kD);
  cvtW(kf_W2, kD, kD, kD);
  gemm(kf_b2, nullptr, 0, kfe, kT, kD, kD, 0, 0);
  { int n = kBT * kD;
    add_bcast_k<<<(n + 255) / 256, 256, 0, stream>>>(x, kfe); }

  // ---- relative position embedding ----
  { int n = kREL * kDH;
    relh_k<<<(n + 255) / 256, 256, 0, stream>>>(rp_W1, rp_b1, relh);
    rel2_k<<<(n + 255) / 256, 256, 0, stream>>>(relh, rp_W2, rp_b2, relE); }

  // ---- transformer layers ----
  for (int l = 0; l < kNL; ++l) {
    layernorm_k<<<kBT, 256, 0, stream>>>(x, att_ln_g + (size_t)l * kD,
                                         att_ln_b + (size_t)l * kD, h, kD);
    cvtR(h, kBT, kD, kD);
    cvtW(att_Wq + (size_t)l * kD * kD, kD, kD, kD);
    gemm(att_bq + (size_t)l * kD, nullptr, 0, qb, kBT, kD, kD, 0, 0);
    cvtW(att_Wk + (size_t)l * kD * kD, kD, kD, kD);
    gemm(att_bk + (size_t)l * kD, nullptr, 0, kb, kBT, kD, kD, 0, 0);
    cvtW(att_Wv + (size_t)l * kD * kD, kD, kD, kD);
    gemm(att_bv + (size_t)l * kD, nullptr, 0, vb, kBT, kD, kD, 0, 0);

    attention_k<<<dim3(kB * kNH, kT / 8), 256, 0, stream>>>(qb, kb, vb, relE, ob);

    cvtR(ob, kBT, kD, kD);
    cvtW(att_Wo + (size_t)l * kD * kD, kD, kD, kD);
    gemm(att_bo + (size_t)l * kD, nullptr, 0, x, kBT, kD, kD, 0, 1);   // residual

    layernorm_k<<<kBT, 256, 0, stream>>>(x, pff_ln_g + (size_t)l * kD,
                                         pff_ln_b + (size_t)l * kD, h, kD);
    cvtR(h, kBT, kD, kD);
    cvtW(pff_W1 + (size_t)l * kD * kDFF, kD, kDFF, kD);
    gemm(pff_b1 + (size_t)l * kDFF, nullptr, 0, big, kBT, kDFF, kD, 2, 0);  // ReLU
    cvtR(big, kBT, kDFF, kDFF);
    cvtW(pff_W2 + (size_t)l * kDFF * kD, kDFF, kD, kDFF);
    gemm(pff_b2 + (size_t)l * kD, nullptr, 0, x, kBT, kD, kDFF, 0, 1);  // residual
  }

  // ---- final LN (in-place) ----
  layernorm_k<<<kBT, 256, 0, stream>>>(x, fin_ln_g, fin_ln_b, x, kD);

  // ---- phase decoder ----
  cvtR(x, kBT, kD, kD);
  cvtW(pd_W1, kD, kHD, kD);
  gemm(pd_b1, nullptr, 0, y, kBT, kHD, kD, 1, 0);
  cvtR(y, kBT, kHD, kHD);
  cvtW(pd_W2, kHD, kP, kHD);
  gemm(pd_b2, nullptr, 0, phase_out, kBT, kP, kHD, 0, 0);

  // ---- gating network ----
  cvtR(phase_out, kBT, kP, 32);
  cvtW(g_W1, kP, kHG, 32);
  gemm(g_b1, nullptr, 0, gg1, kBT, kHG, 32, 1, 0);
  cvtR(gg1, kBT, kHG, kHG);
  cvtW(g_W2, kHG, kHG, kHG);
  gemm(g_b2, nullptr, 0, gg2, kBT, kHG, kHG, 1, 0);
  cvtR(gg2, kBT, kHG, kHG);
  cvtW(g_W3, kHG, kE, kHG);
  gemm(g_b3, nullptr, 0, glog, kBT, kE, kHG, 0, 0);
  softmax_rows_k<<<(kBT + 255) / 256, 256, 0, stream>>>(glog, gate, kBT);

  // ---- MoE decoder: y = sum_e gate_e * (x @ W_e + b_e), blended in-epilogue ----
  cvtR(x, kBT, kD, kD);
  for (int e = 0; e < kE; ++e) {
    cvtW(md_W0 + (size_t)e * kD * kHD, kD, kHD, kD);
    gemm(md_b0 + (size_t)e * kHD, gate + e, kE, y, kBT, kHD, kD, 0, e ? 1 : 0);
  }
  prelu_k<<<(kBT * kHD + 255) / 256, 256, 0, stream>>>(y, kBT * kHD);

  cvtR(y, kBT, kHD, kHD);
  for (int e = 0; e < kE; ++e) {
    cvtW(md_W1 + (size_t)e * kHD * kHD, kHD, kHD, kHD);
    gemm(md_b1 + (size_t)e * kHD, gate + e, kE, big, kBT, kHD, kHD, 0, e ? 1 : 0);
  }
  prelu_k<<<(kBT * kHD + 255) / 256, 256, 0, stream>>>(big, kBT * kHD);

  cvtR(big, kBT, kHD, kHD);
  for (int e = 0; e < kE; ++e) {
    cvtW(md_W2 + (size_t)e * kHD * kM, kHD, kM, kHD);
    gemm(md_b2 + (size_t)e * kM, gate + e, kE, yout, kBT, kM, kHD, 0, e ? 1 : 0);
  }
}
